// PGBF_12189117186116
// MI455X (gfx1250) — matrix operand order repaired and checked by reading, not compiled, not run on hardware
//
#include <hip/hip_runtime.h>
#include <stdint.h>

// ---- problem constants -----------------------------------------------------
#define NP   8192     // WSI patches
#define DIN  384
#define DHID 512
#define HSNN 256
#define KTOP 6

typedef __attribute__((ext_vector_type(16))) _Float16 v16h;
typedef __attribute__((ext_vector_type(8)))  _Float16 v8h;
typedef __attribute__((ext_vector_type(8)))  float    v8f;

__device__ __forceinline__ float leakyf(float x) { return x >= 0.f ? x : 0.01f * x; }
__device__ __forceinline__ float eluf(float x)   { return x > 0.f ? x : expm1f(x); }

// ---------------------------------------------------------------------------
// CDNA5 async global->LDS staging (ASYNCcnt path, bypasses VGPRs)
// LDS address is taken from the VDST VGPR; generic LDS pointer low 32 bits = LDS offset
// ---------------------------------------------------------------------------
__device__ __forceinline__ void async_copy_b128(void* lds_dst, const void* gsrc)
{
    uint32_t lds_off = (uint32_t)(uintptr_t)lds_dst;
    uint64_t gaddr   = (uint64_t)(uintptr_t)gsrc;
    asm volatile("global_load_async_to_lds_b128 %0, %1, off"
                 :: "v"(lds_off), "v"(gaddr) : "memory");
}
// async "done" returns in order: cnt<=3 tolerates the 3 in-flight issues of
// the next (double-buffered) tile while guaranteeing the current tile landed.
__device__ __forceinline__ void wait_async_le3()
{
    asm volatile("s_wait_asynccnt 0x3" ::: "memory");
}
__device__ __forceinline__ void wait_async_le0()
{
    asm volatile("s_wait_asynccnt 0x0" ::: "memory");
}

// load a 16x32 f16 A-fragment (wave32, contiguous k order) from an LDS row:
// two contiguous 8-half (16B) chunks at +g*16 and +g*16+8
__device__ __forceinline__ v16h load_a_frag(const _Float16* rowp, int g)
{
    v8h lo = *(const v8h*)(rowp + g * 16);
    v8h hi = *(const v8h*)(rowp + g * 16 + 8);
    return __builtin_shufflevector(lo, hi, 0,1,2,3,4,5,6,7,8,9,10,11,12,13,14,15);
}

// stage one K-step tile pair (A 64x32, W 128x32, f16) with async B128 copies:
// 256 + 512 = 768 segments -> exactly 3 async issues per thread, wave-uniform
__device__ __forceinline__ void stage_tiles(const _Float16* __restrict__ A16,
                                            const _Float16* __restrict__ W16,
                                            int K, int mBase, int nBase,
                                            int tid, int k0,
                                            _Float16* Asb, _Float16* Wsb)
{
    int rowA = tid >> 2, segA = tid & 3;
    async_copy_b128(Asb + rowA * 32 + segA * 8,
                    A16 + (mBase + rowA) * K + k0 + segA * 8);
    #pragma unroll
    for (int i = 0; i < 2; ++i) {
        int e = tid + i * 256;
        int row = e >> 2, seg = e & 3;
        async_copy_b128(Wsb + row * 32 + seg * 8,
                        W16 + (nBase + row) * K + k0 + seg * 8);
    }
}

// ---------------------------------------------------------------------------
// f32 -> f16 prep pass:  dst = (a [op] b) * scale ;  op: 0=copy, 1=add, 2=mul
// ---------------------------------------------------------------------------
__global__ void prep16_kernel(const float* __restrict__ a,
                              const float* __restrict__ b,
                              _Float16* __restrict__ dst, int mode, float scale)
{
    int i = blockIdx.x * 256 + threadIdx.x;
    float v = a[i];
    if (mode == 1)      v += b[i];
    else if (mode == 2) v *= b[i];
    dst[i] = (_Float16)(v * scale);
}

// ---------------------------------------------------------------------------
// Six SNN omic branches: h = elu(W1 x + b1); out = elu(W2 h + b2)
// ---------------------------------------------------------------------------
__global__ void snn_kernel(const float* x1, const float* x2, const float* x3,
                           const float* x4, const float* x5, const float* x6,
                           const float* w11, const float* w12, const float* w13,
                           const float* w14, const float* w15, const float* w16,
                           const float* __restrict__ sig_b1,
                           const float* __restrict__ sig_w2,
                           const float* __restrict__ sig_b2,
                           float* __restrict__ out)
{
    __shared__ float xb[1536];
    __shared__ float h1[HSNN];
    const int sizes[6] = {89, 334, 534, 471, 1510, 482};
    const float* xs[6] = {x1, x2, x3, x4, x5, x6};
    const float* ws[6] = {w11, w12, w13, w14, w15, w16};
    int br = blockIdx.x, tid = threadIdx.x;
    int s = sizes[br];
    for (int j = tid; j < s; j += 256) xb[j] = xs[br][j];
    __syncthreads();
    const float* W1 = ws[br];
    float acc = sig_b1[br * HSNN + tid];
    for (int j = 0; j < s; ++j) acc += W1[tid * s + j] * xb[j];
    h1[tid] = eluf(acc);
    __syncthreads();
    const float* W2 = sig_w2 + br * HSNN * HSNN;
    float acc2 = sig_b2[br * HSNN + tid];
    for (int j = 0; j < HSNN; ++j) acc2 += W2[tid * HSNN + j] * h1[j];
    out[br * HSNN + tid] = eluf(acc2);
}

// ---------------------------------------------------------------------------
// WMMA GEMM, all-f16 operands:  out[M x N] = act( A16[M x K] @ W16[N x K]^T + bias )
// Block = 256 threads (8 waves, 4x2 wave grid), block tile 64x128; each wave
// owns a 16x64 strip (4 accumulators) -> one A fragment feeds 4 v_wmma/K-step.
// Double-buffered async global->LDS staging: tile s+1 is issued before the
// wait for tile s (s_wait_asynccnt 0x3), hiding global latency behind WMMA.
// Optional secondary f16 output (value * scale16) for downstream WMMA use.
// grid = (N/128, M/64); requires N%128==0, M%64==0, K%32==0.
// ---------------------------------------------------------------------------
__global__ void gemm_wmma16_kernel(const _Float16* __restrict__ A16,
                                   const _Float16* __restrict__ W16,
                                   const float* __restrict__ bias,
                                   float* __restrict__ out,
                                   _Float16* __restrict__ out16,
                                   int K, int Nld, float scale16,
                                   int act, int out_add)
{
    int tid = threadIdx.x, lane = tid & 31, wave = tid >> 5;
    int wr = wave >> 1, wc = wave & 1;               // 4x2 wave grid
    int mBase = blockIdx.y * 64, nBase = blockIdx.x * 128;

    __shared__ __align__(32) _Float16 As[2][64 * 32];    // 2 x 4 KB
    __shared__ __align__(32) _Float16 Ws[2][128 * 32];   // 2 x 8 KB

    v8f acc0 = {}, acc1 = {}, acc2v = {}, acc3 = {};
    int g = lane >> 4, mloc = lane & 15;
    int nsteps = K / 32;

    stage_tiles(A16, W16, K, mBase, nBase, tid, 0, As[0], Ws[0]);

    for (int s = 0; s < nsteps; ++s) {
        int p = s & 1;
        if (s + 1 < nsteps) {
            stage_tiles(A16, W16, K, mBase, nBase, tid, (s + 1) * 32,
                        As[1 - p], Ws[1 - p]);
            wait_async_le3();
        } else {
            wait_async_le0();
        }
        __syncthreads();

        v16h a = load_a_frag(&As[p][(wr * 16 + mloc) * 32], g);
        #pragma unroll
        for (int t = 0; t < 4; ++t) {
            int nl = wc * 64 + t * 16 + mloc;        // local output column
            v16h b = *(const v16h*)&Ws[p][nl * 32 + g * 16];   // 32B aligned
            if (t == 0) acc0 = __builtin_amdgcn_wmma_f32_16x16x32_f16(false, a, false, b, (short)0, acc0, false, false);
            if (t == 1) acc1 = __builtin_amdgcn_wmma_f32_16x16x32_f16(false, a, false, b, (short)0, acc1, false, false);
            if (t == 2) acc2v = __builtin_amdgcn_wmma_f32_16x16x32_f16(false, a, false, b, (short)0, acc2v, false, false);
            if (t == 3) acc3 = __builtin_amdgcn_wmma_f32_16x16x32_f16(false, a, false, b, (short)0, acc3, false, false);
        }
        __syncthreads();   // all waves done with buf p before it is re-staged
    }

    #pragma unroll
    for (int t = 0; t < 4; ++t) {
        v8f acc = (t == 0) ? acc0 : (t == 1) ? acc1 : (t == 2) ? acc2v : acc3;
        int n = nBase + wc * 64 + t * 16 + mloc;
        float bv = bias ? bias[n] : 0.f;
        #pragma unroll
        for (int r = 0; r < 8; ++r) {                // C/D: m = g*8+r, n = lane&15
            int m = mBase + wr * 16 + g * 8 + r;
            float v = acc[r] + bv;
            if (act == 1) v = leakyf(v);
            int oi = m * Nld + n;
            if (out_add) out[oi] += v; else out[oi] = v;
            if (out16) out16[oi] = (_Float16)(v * scale16);
        }
    }
}

// ---------------------------------------------------------------------------
// Fused logits GEMM + top-k(6):  logits[m,n] = dot(e_h[m,:], e_t[n,:]) / sqrt(DH)
// Inputs are pre-converted f16 (scale folded into ehs16). One block per
// 16-row strip; 8 waves sweep 8192 columns in 16x16 WMMA tiles; per-lane
// running top-6 in registers; final LDS merge. The 8192x8192 matrix is never
// materialized (256MB > 192MB L2).
// ---------------------------------------------------------------------------
__global__ void logits_topk_kernel(const _Float16* __restrict__ ehs16,
                                   const _Float16* __restrict__ et16,
                                   float* __restrict__ topv,
                                   int* __restrict__ topi)
{
    int tid = threadIdx.x, lane = tid & 31, wave = tid >> 5;
    int mBase = blockIdx.x * 16;

    __shared__ __align__(32) _Float16 Ast[16 * 512]; // A row-strip, 16 KB
    __shared__ float tile[8][16][16];
    __shared__ float candV[8][32][6];
    __shared__ int   candI[8][32][6];

    // async-stage the whole 16x512 f16 strip (1024 B128 segments, 4/thread)
    #pragma unroll
    for (int i = 0; i < 4; ++i) {
        int e = tid + i * 256;
        int row = e >> 6, seg = e & 63;
        async_copy_b128(&Ast[row * 512 + seg * 8],
                        &ehs16[(mBase + row) * DHID + seg * 8]);
    }
    wait_async_le0();
    __syncthreads();

    int g = lane >> 4, mloc = lane & 15;
    float tv[6]; int ti[6];
    #pragma unroll
    for (int j = 0; j < 6; ++j) { tv[j] = -3.4e38f; ti[j] = 0; }

    for (int it = 0; it < NP / 16 / 8; ++it) {  // 64 column tiles per wave
        int ct = wave + it * 8;
        v8f acc = {};
        for (int kt = 0; kt < DHID / 32; ++kt) {
            v16h a = load_a_frag(&Ast[mloc * 512 + kt * 32], g);
            v16h b = *(const v16h*)&et16[(ct * 16 + mloc) * DHID + kt * 32 + g * 16];
            acc = __builtin_amdgcn_wmma_f32_16x16x32_f16(false, a, false, b,
                                                         (short)0, acc, false, false);
        }
        #pragma unroll
        for (int r = 0; r < 8; ++r)
            tile[wave][g * 8 + r][mloc] = acc[r];
        __syncthreads();
        #pragma unroll
        for (int c = 0; c < 8; ++c) {            // lane: row=mloc, cols g*8..g*8+7
            float v = tile[wave][mloc][g * 8 + c];
            int col = ct * 16 + g * 8 + c;
            if (v > tv[5]) {
                int p = 5;
                while (p > 0 && v > tv[p - 1]) { tv[p] = tv[p-1]; ti[p] = ti[p-1]; --p; }
                tv[p] = v; ti[p] = col;
            }
        }
        __syncthreads();
    }

    #pragma unroll
    for (int j = 0; j < 6; ++j) { candV[wave][lane][j] = tv[j]; candI[wave][lane][j] = ti[j]; }
    __syncthreads();

    if (tid < 16) {                              // merge 8 waves x 2 half-lanes x 6
        int row = mBase + tid;
        for (int sel = 0; sel < 6; ++sel) {
            float best = -3.4e38f; int bw = 0, bl = tid, bj = 0;
            for (int w = 0; w < 8; ++w)
                for (int hh = 0; hh < 2; ++hh) {
                    int l = tid + hh * 16;
                    for (int j = 0; j < 6; ++j) {
                        float v = candV[w][l][j];
                        if (v > best) { best = v; bw = w; bl = l; bj = j; }
                    }
                }
            topv[row * 6 + sel] = best;
            topi[row * 6 + sel] = candI[bw][bl][bj];
            candV[bw][bl][bj] = -3.4e38f;
        }
    }
}

// ---------------------------------------------------------------------------
// helpers: block reductions (wave32 shuffles + LDS across 8 waves)
// ---------------------------------------------------------------------------
__device__ __forceinline__ float block_reduce_sum(float v, float* sm)
{
    int tid = threadIdx.x;
    #pragma unroll
    for (int off = 16; off > 0; off >>= 1) v += __shfl_down(v, off, 32);
    if ((tid & 31) == 0) sm[tid >> 5] = v;
    __syncthreads();
    float tot = 0.f;
    #pragma unroll
    for (int w = 0; w < 8; ++w) tot += sm[w];
    __syncthreads();
    return tot;
}

// ---------------------------------------------------------------------------
// Neighbor aggregation: softmax(top6), gather Nb=e_t[idx], tanh gate,
// ka = sum(Nb)*sum(gate), kp = softmax(ka), e_Nh = sum_k kp*Nb.
// ---------------------------------------------------------------------------
__global__ void neighbor_kernel(const float* __restrict__ e_h,
                                const float* __restrict__ e_t,
                                const float* __restrict__ topv,
                                const int* __restrict__ topi,
                                float* __restrict__ eNh)
{
    __shared__ float sm[8];
    int row = blockIdx.x, tid = threadIdx.x;
    float pv[6]; int idx[6];
    float mx = topv[row * 6];            // sorted descending -> [0] is max
    float ps = 0.f;
    #pragma unroll
    for (int k = 0; k < 6; ++k) {
        idx[k] = topi[row * 6 + k];
        pv[k] = expf(topv[row * 6 + k] - mx);
        ps += pv[k];
    }
    float inv = 1.f / ps;
    #pragma unroll
    for (int k = 0; k < 6; ++k) pv[k] *= inv;

    float s1[6] = {0,0,0,0,0,0}, s2[6] = {0,0,0,0,0,0};
    for (int d = tid; d < DHID; d += 256) {
        float eh = e_h[row * DHID + d];
        #pragma unroll
        for (int k = 0; k < 6; ++k) {
            float nb  = e_t[idx[k] * DHID + d];
            float ehr = pv[k] * nb + (1.f - pv[k]) * eh;
            float gt  = tanhf(eh + ehr);
            s1[k] += nb; s2[k] += gt;
        }
    }
    float ka[6];
    for (int k = 0; k < 6; ++k) {
        float r1 = block_reduce_sum(s1[k], sm);
        float r2 = block_reduce_sum(s2[k], sm);
        ka[k] = r1 * r2;
    }
    float km = ka[0];
    #pragma unroll
    for (int k = 1; k < 6; ++k) km = fmaxf(km, ka[k]);
    float kp[6], ks = 0.f;
    #pragma unroll
    for (int k = 0; k < 6; ++k) { kp[k] = expf(ka[k] - km); ks += kp[k]; }
    float kinv = 1.f / ks;
    for (int d = tid; d < DHID; d += 256) {
        float o = 0.f;
        #pragma unroll
        for (int k = 0; k < 6; ++k) o += kp[k] * kinv * e_t[idx[k] * DHID + d];
        eNh[row * DHID + d] = o;
    }
}

// ---------------------------------------------------------------------------
// small elementwise / reduction kernels
// ---------------------------------------------------------------------------
__global__ void colmean_kernel(const float* __restrict__ h, float* __restrict__ cm)
{
    int c = blockIdx.x * 256 + threadIdx.x;      // 512 columns, 2 blocks
    float s = 0.f;
    for (int m = 0; m < NP; ++m) s += h[m * DHID + c];
    cm[c] = s * (1.f / (float)NP);
}

__global__ void center_kernel(float* __restrict__ h, const float* __restrict__ cm,
                              _Float16* __restrict__ h16)
{
    int i = blockIdx.x * 256 + threadIdx.x;      // NP*DHID elements
    float v = (h[i] + cm[i & (DHID - 1)]) * 0.5f;
    h[i] = v;
    h16[i] = (_Float16)v;
}

__global__ void attn_g_kernel(const float* __restrict__ a1,
                              const float* __restrict__ w2,
                              const float* __restrict__ b2,
                              float* __restrict__ g)
{
    int row = blockIdx.x * 256 + threadIdx.x;
    float s = b2[0];
    for (int d = 0; d < DHID / 2; ++d) s += a1[row * (DHID / 2) + d] * w2[d];
    g[row] = s;
}

__global__ void softmax_g_kernel(float* __restrict__ g)
{
    __shared__ float sm[8];
    int tid = threadIdx.x;
    float mx = -3.4e38f;
    for (int n = tid; n < NP; n += 256) mx = fmaxf(mx, g[n]);
    #pragma unroll
    for (int off = 16; off > 0; off >>= 1) mx = fmaxf(mx, __shfl_down(mx, off, 32));
    if ((tid & 31) == 0) sm[tid >> 5] = mx;
    __syncthreads();
    float m = sm[0];
    #pragma unroll
    for (int w = 1; w < 8; ++w) m = fmaxf(m, sm[w]);
    __syncthreads();
    float ssum = 0.f;
    for (int n = tid; n < NP; n += 256) { float e = expf(g[n] - m); g[n] = e; ssum += e; }
    float tot = block_reduce_sum(ssum, sm);
    float inv = 1.f / tot;
    for (int n = tid; n < NP; n += 256) g[n] *= inv;
}

__global__ void readout_kernel(const float* __restrict__ a,
                               const float* __restrict__ x,
                               float* __restrict__ eg)
{
    int d = blockIdx.x * 256 + threadIdx.x;      // 512 dims, 2 blocks
    float s = 0.f;
    for (int n = 0; n < NP; ++n) s += a[n] * x[n * DHID + d];
    eg[d] = s;
}

// ---------------------------------------------------------------------------
// launch
// ---------------------------------------------------------------------------
extern "C" void kernel_launch(void* const* d_in, const int* in_sizes, int n_in,
                              void* d_out, int out_size, void* d_ws, size_t ws_size,
                              hipStream_t stream)
{
    (void)in_sizes; (void)n_in; (void)out_size; (void)ws_size;

    // dict order: (x_omic_i, sig_w1_i) x6 interleaved, then the rest
    const float* xo[6]; const float* w1[6];
    for (int i = 0; i < 6; ++i) { xo[i] = (const float*)d_in[2*i]; w1[i] = (const float*)d_in[2*i+1]; }
    const float* x_path = (const float*)d_in[12];
    const float* sig_b1 = (const float*)d_in[13];
    const float* sig_w2 = (const float*)d_in[14];
    const float* sig_b2 = (const float*)d_in[15];
    const float* fc1_w  = (const float*)d_in[16];
    const float* fc1_b  = (const float*)d_in[17];
    const float* wh_w   = (const float*)d_in[18];
    const float* wh_b   = (const float*)d_in[19];
    const float* wt_w   = (const float*)d_in[20];
    const float* wt_b   = (const float*)d_in[21];
    const float* lin1_w = (const float*)d_in[22];
    const float* lin1_b = (const float*)d_in[23];
    const float* lin2_w = (const float*)d_in[24];
    const float* lin2_b = (const float*)d_in[25];
    const float* att_w1 = (const float*)d_in[26];
    const float* att_b1 = (const float*)d_in[27];
    const float* att_w2 = (const float*)d_in[28];
    const float* att_b2 = (const float*)d_in[29];

    // ---- workspace layout (f32 region, element offsets) --------------------
    float* ws   = (float*)d_ws;
    float* h    = ws;                         // NP*DHID   (reused as e_Nh later)
    float* e_h  = ws + 4194304;               // NP*DHID
    float* e_t  = ws + 8388608;               // NP*DHID
    float* a1   = ws + 12582912;              // NP*256
    float* topv = ws + 14680064;              // NP*6
    int*   topi = (int*)(ws + 14729216);      // NP*6
    float* colm = ws + 14778368;              // 512
    float* gbuf = ws + 14778880;              // NP
    // ---- f16 region (half offsets from hb) ---------------------------------
    _Float16* hb      = (_Float16*)(ws + 14787072);
    _Float16* x16     = hb;                   // NP*DIN   = 3145728
    _Float16* h16     = hb + 3145728;         // NP*DHID  = 4194304
    _Float16* ehs16   = hb + 7340032;         // NP*DHID (scaled e_h)
    _Float16* et16    = hb + 11534336;        // NP*DHID
    _Float16* comb16  = hb + 15728640;        // NP*DHID (lin1 then lin2 input)
    _Float16* eh2_16  = hb + 19922944;        // NP*DHID
    _Float16* fc1w16  = hb + 24117248;        // 512*384
    _Float16* whw16   = hb + 24313856;        // 512*512
    _Float16* wtw16   = hb + 24576000;        // 512*512
    _Float16* l1w16   = hb + 24838144;        // 512*512
    _Float16* l2w16   = hb + 25100288;        // 512*512
    _Float16* a1w16   = hb + 25362432;        // 256*512

    float* out       = (float*)d_out;
    float* out_eomic = out;                   // [6,1,256]
    float* out_eh2   = out + 6 * HSNN;        // [1,NP,DHID]
    float* out_eg    = out + 6 * HSNN + NP * DHID;  // [1,DHID]

    const float RSQRT_DH = 0.044194173824159216f;

    // 1) omic SNN branches
    snn_kernel<<<6, 256, 0, stream>>>(xo[0], xo[1], xo[2], xo[3], xo[4], xo[5],
                                      w1[0], w1[1], w1[2], w1[3], w1[4], w1[5],
                                      sig_b1, sig_w2, sig_b2, out_eomic);

    // 2) one-time f16 conversions (weights + x_path)
    prep16_kernel<<<(NP*DIN)/256, 256, 0, stream>>>(x_path, nullptr, x16, 0, 1.f);
    prep16_kernel<<<(DHID*DIN)/256, 256, 0, stream>>>(fc1_w, nullptr, fc1w16, 0, 1.f);
    prep16_kernel<<<(DHID*DHID)/256, 256, 0, stream>>>(wh_w, nullptr, whw16, 0, 1.f);
    prep16_kernel<<<(DHID*DHID)/256, 256, 0, stream>>>(wt_w, nullptr, wtw16, 0, 1.f);
    prep16_kernel<<<(DHID*DHID)/256, 256, 0, stream>>>(lin1_w, nullptr, l1w16, 0, 1.f);
    prep16_kernel<<<(DHID*DHID)/256, 256, 0, stream>>>(lin2_w, nullptr, l2w16, 0, 1.f);
    prep16_kernel<<<((DHID/2)*DHID)/256, 256, 0, stream>>>(att_w1, nullptr, a1w16, 0, 1.f);

    // 3) h = leaky(x_path @ fc1_w.T + fc1_b)      [NP x DHID], K=384
    gemm_wmma16_kernel<<<dim3(DHID/128, NP/64), 256, 0, stream>>>(
        x16, fc1w16, fc1_b, h, nullptr, DIN, DHID, 1.f, 1, 0);

    // 4) h = (h + colmean(h)) * 0.5   (also emits f16 copy)
    colmean_kernel<<<2, 256, 0, stream>>>(h, colm);
    center_kernel<<<(NP * DHID) / 256, 256, 0, stream>>>(h, colm, h16);

    // 5) e_h / e_t  (epilogues emit f16 copies; scale folded into ehs16)
    gemm_wmma16_kernel<<<dim3(DHID/128, NP/64), 256, 0, stream>>>(
        h16, whw16, wh_b, e_h, ehs16, DHID, DHID, RSQRT_DH, 0, 0);
    gemm_wmma16_kernel<<<dim3(DHID/128, NP/64), 256, 0, stream>>>(
        h16, wtw16, wt_b, e_t, et16, DHID, DHID, 1.f, 0, 0);

    // 6) fused logits GEMM + top-k(6)  (8192x8192x512, never materialized)
    logits_topk_kernel<<<NP / 16, 256, 0, stream>>>(ehs16, et16, topv, topi);

    // 7) neighbor aggregation -> e_Nh (reuses h's f32 region)
    neighbor_kernel<<<NP, 256, 0, stream>>>(e_h, e_t, topv, topi, h);

    // 8) e_h2 = leaky((e_h+e_Nh)@lin1.T+b1) + leaky((e_h*e_Nh)@lin2.T+b2)
    prep16_kernel<<<(NP*DHID)/256, 256, 0, stream>>>(e_h, h, comb16, 1, 1.f);
    gemm_wmma16_kernel<<<dim3(DHID/128, NP/64), 256, 0, stream>>>(
        comb16, l1w16, lin1_b, out_eh2, nullptr, DHID, DHID, 1.f, 1, 0);
    prep16_kernel<<<(NP*DHID)/256, 256, 0, stream>>>(e_h, h, comb16, 2, 1.f);
    gemm_wmma16_kernel<<<dim3(DHID/128, NP/64), 256, 0, stream>>>(
        comb16, l2w16, lin2_b, out_eh2, nullptr, DHID, DHID, 1.f, 1, 1);

    // 9) global attention readout
    prep16_kernel<<<(NP*DHID)/256, 256, 0, stream>>>(out_eh2, nullptr, eh2_16, 0, 1.f);
    gemm_wmma16_kernel<<<dim3((DHID/2)/128, NP/64), 256, 0, stream>>>(
        eh2_16, a1w16, att_b1, a1, nullptr, DHID, DHID/2, 1.f, 1, 0);
    attn_g_kernel<<<NP / 256, 256, 0, stream>>>(a1, att_w2, att_b2, gbuf);
    softmax_g_kernel<<<1, 256, 0, stream>>>(gbuf);
    readout_kernel<<<2, 256, 0, stream>>>(gbuf, out_eh2, out_eg);
}
